// Space_attention_42880953483835
// MI455X (gfx1250) — matrix operand order repaired and checked by reading, not compiled, not run on hardware
//
#include <hip/hip_runtime.h>
#include <hip/hip_bf16.h>

// ---------------------------------------------------------------------------
// Space attention for MI455X (gfx1250, wave32, WMMA f16 16x16x32).
// B=8, C=64, H=W=64, N=4096.
// Fused flash-attention: never materializes the 512MB score tensor.
// ---------------------------------------------------------------------------

typedef _Float16 half_t;
typedef __attribute__((ext_vector_type(16))) _Float16 v16h;
typedef __attribute__((ext_vector_type(8)))  _Float16 h8;
typedef __attribute__((ext_vector_type(8)))  float    v8f;

union V16u { v16h v; h8 h[2]; };

__device__ __forceinline__ v8f wmma16(v16h a, v16h b, v8f c) {
  // (neg_a, A, neg_b, B, c_mod, C, reuse_a, reuse_b)
  return __builtin_amdgcn_wmma_f32_16x16x32_f16(false, a, false, b, (short)0, c,
                                                false, false);
}

// A fragment (16x32 f16): lane holds row r = lane%16.
// halves i -> K = (lane<16) ? {0..15} : {16..31}.
// rowp points at K=0 of this lane's row; a0 = (lane<16 ? 0 : 8).
__device__ __forceinline__ v16h loadA(const half_t* rowp, int a0) {
  V16u u;
  u.h[0] = *(const h8*)(rowp + 2 * a0);
  u.h[1] = *(const h8*)(rowp + 2 * a0 + 8);
  return u.v;
}

// B fragment (32x16 f16): lane holds col c = lane%16.
// halves i -> K = i + (lane<16 ? 0 : 16). p already offset to this lane's K base.
__device__ __forceinline__ v16h loadB(const half_t* p) {
  V16u u;
  u.h[0] = *(const h8*)(p);
  u.h[1] = *(const h8*)(p + 8);
  return u.v;
}

// B fragment sourced from 16 contiguous f32 (weights), converted to f16.
__device__ __forceinline__ v16h loadB_f32(const float* p) {
  v16h r;
#pragma unroll
  for (int i = 0; i < 16; ++i) r[i] = (half_t)p[i];
  return r;
}

// ---------------------------------------------------------------------------
// Kernel 1: K/Q/V projections.  x[b][c][n] -> Kt[b][n][o], Qt[b][n][o] (f16,
// transposed layouts so attention fragments load contiguously) and V[b][o][n].
// Grid: 8 batches * 64 n-chunks = 512 blocks of 256 threads (8 waves).
// ---------------------------------------------------------------------------
__global__ __launch_bounds__(256) void proj_kernel(
    const float* __restrict__ x,
    const float* __restrict__ Wk, const float* __restrict__ bk,
    const float* __restrict__ Wq, const float* __restrict__ bq,
    const float* __restrict__ Wv, const float* __restrict__ bv,
    half_t* __restrict__ Kt, half_t* __restrict__ Qt, half_t* __restrict__ Vn)
{
  __shared__ __align__(16) half_t xT[64 * 72];   // [n][c], pad 72 halves
  const int b  = blockIdx.x >> 6;
  const int n0 = (blockIdx.x & 63) << 6;
  const float* xb = x + ((size_t)b << 18);       // 64*4096 floats per batch
  const int t = threadIdx.x;
#pragma unroll
  for (int it = 0; it < 16; ++it) {
    int idx = t + it * 256;
    int c = idx >> 6, n = idx & 63;              // coalesced read, transposed store
    xT[n * 72 + c] = (half_t)xb[((size_t)c << 12) + n0 + n];
  }
  __syncthreads();

  const int wave = t >> 5, lane = t & 31;
  const int r    = lane & 15;
  const int ho   = (lane < 16) ? 0 : 8;
  const int bo16 = (lane < 16) ? 0 : 16;

  // A fragments (rows = n, contraction = c), reused for all 12 output tiles
  const half_t* arow = &xT[(wave * 16 + r) * 72];
  const v16h a0 = loadA(arow, ho);
  const v16h a1 = loadA(arow + 32, ho);

  const float* Ws[3] = {Wk, Wq, Wv};
  const float* Bs[3] = {bk, bq, bv};
#pragma unroll
  for (int pj = 0; pj < 3; ++pj) {
    const float* W    = Ws[pj];
    const float* bias = Bs[pj];
#pragma unroll
    for (int ot = 0; ot < 4; ++ot) {
      const int c = ot * 16 + r;                 // output channel o
      v16h b0 = loadB_f32(W + c * 64 + bo16);        // B[k=c][col=o] = W[o][c]
      v16h b1 = loadB_f32(W + c * 64 + 32 + bo16);
      v8f acc = {};
      acc = wmma16(a0, b0, acc);
      acc = wmma16(a1, b1, acc);
      const float bia = bias[c];
      if (pj < 2) {                              // Kt / Qt: [n][o] layout
        half_t* dst = (pj == 0 ? Kt : Qt) + ((size_t)b << 18);
        const int nbase = n0 + wave * 16 + ho;
#pragma unroll
        for (int j = 0; j < 8; ++j)
          dst[(size_t)(nbase + j) * 64 + c] = (half_t)(acc[j] + bia);
      } else {                                   // V: natural [o][n] layout
        half_t* dst = Vn + ((size_t)b << 18) + ((size_t)c << 12)
                    + n0 + wave * 16 + ho;
        h8 pk;
#pragma unroll
        for (int j = 0; j < 8; ++j) pk[j] = (half_t)(acc[j] + bia);
        *(h8*)dst = pk;
      }
    }
  }
}

// ---------------------------------------------------------------------------
// Kernel 2: flash attention.  One wave owns a 16-row tile of scores; loops
// over all 4096 key columns in chunks of 128 with online softmax.
// vec[b][o][n] (f32) is produced, already divided by the softmax denominator.
// Grid: 2048 row-tiles / 8 waves = 256 blocks of 256 threads.
// ---------------------------------------------------------------------------
__global__ __launch_bounds__(256) void attn_kernel(
    const half_t* __restrict__ Kt, const half_t* __restrict__ Qt,
    const half_t* __restrict__ Vn, float* __restrict__ vec)
{
  __shared__ __align__(16) half_t plds[8 * 16 * 40];  // per-wave 16x32 P tile, pad 40
  const int t    = threadIdx.x;
  const int wave = t >> 5, lane = t & 31;
  const int tile = blockIdx.x * 8 + wave;
  const int b    = tile >> 8;
  const int n0   = (tile & 255) << 4;
  const half_t* Ktb = Kt + ((size_t)b << 18);
  const half_t* Qtb = Qt + ((size_t)b << 18);
  const half_t* Vb  = Vn + ((size_t)b << 18);
  const int r    = lane & 15;
  const int ho   = (lane < 16) ? 0 : 8;
  const int bo16 = (lane < 16) ? 0 : 16;

  // K row fragments for this n-tile (rows = n, contraction = o), loaded once.
  const half_t* krow = Ktb + (size_t)(n0 + r) * 64;
  const v16h ka0 = loadA(krow, ho);
  const v16h ka1 = loadA(krow + 32, ho);
  half_t* pw = &plds[wave * 16 * 40];

  v8f o0 = {}, o1 = {}, o2 = {}, o3 = {};
  float mrow[8], lrow[8];
#pragma unroll
  for (int j = 0; j < 8; ++j) { mrow[j] = -__builtin_inff(); lrow[j] = 0.f; }

  for (int m0 = 0; m0 < 4096; m0 += 128) {
    // ---- S = K^T Q for 16x128 chunk (8 sub-tiles, 2 WMMAs each) ----
    v8f s[8];
#pragma unroll
    for (int ms = 0; ms < 8; ++ms) {
      const half_t* q = Qtb + (size_t)(m0 + ms * 16 + r) * 64;
      v8f acc = {};
      acc = wmma16(ka0, loadB(q + bo16), acc);
      acc = wmma16(ka1, loadB(q + 32 + bo16), acc);
      s[ms] = acc;
    }
    // ---- online softmax: chunk row max (reduce over 8 frags, then 16 lanes)
    float cm[8];
#pragma unroll
    for (int j = 0; j < 8; ++j) {
      float v = s[0][j];
#pragma unroll
      for (int ms = 1; ms < 8; ++ms) v = fmaxf(v, s[ms][j]);
      cm[j] = v;
    }
#pragma unroll
    for (int d = 1; d < 16; d <<= 1)
#pragma unroll
      for (int j = 0; j < 8; ++j) cm[j] = fmaxf(cm[j], __shfl_xor(cm[j], d));

    float corr[8];
#pragma unroll
    for (int j = 0; j < 8; ++j) {
      float mn = fmaxf(mrow[j], cm[j]);
      corr[j] = __expf(mrow[j] - mn);
      mrow[j] = mn;
    }
    // ---- P = exp(S - m), row sums ----
    float cs[8];
#pragma unroll
    for (int j = 0; j < 8; ++j) cs[j] = 0.f;
#pragma unroll
    for (int ms = 0; ms < 8; ++ms)
#pragma unroll
      for (int j = 0; j < 8; ++j) {
        float p = __expf(s[ms][j] - mrow[j]);
        s[ms][j] = p;
        cs[j] += p;
      }
#pragma unroll
    for (int d = 1; d < 16; d <<= 1)
#pragma unroll
      for (int j = 0; j < 8; ++j) cs[j] += __shfl_xor(cs[j], d);
#pragma unroll
    for (int j = 0; j < 8; ++j) lrow[j] = lrow[j] * corr[j] + cs[j];
    // ---- rescale O accumulators ----
#pragma unroll
    for (int j = 0; j < 8; ++j) {
      o0[j] *= corr[j]; o1[j] *= corr[j]; o2[j] *= corr[j]; o3[j] *= corr[j];
    }
    // ---- O += P @ V^T, 32 key columns per step (C-layout -> A-layout via LDS)
#pragma unroll
    for (int mc = 0; mc < 4; ++mc) {
#pragma unroll
      for (int hc = 0; hc < 2; ++hc) {
        const int ms = mc * 2 + hc;
#pragma unroll
        for (int j = 0; j < 8; ++j)
          pw[(ho + j) * 40 + hc * 16 + r] = (half_t)s[ms][j];
      }
      asm volatile("s_wait_dscnt 0x0" ::: "memory");  // cross-lane LDS visibility
      const half_t* prow = pw + r * 40;
      V16u u;
      u.h[0] = *(const h8*)(prow + 2 * ho);
      u.h[1] = *(const h8*)(prow + 2 * ho + 8);
      const v16h pa = u.v;
      const int mb = m0 + mc * 32;
      o0 = wmma16(pa, loadB(Vb + ((size_t)(r)      << 12) + mb + bo16), o0);
      o1 = wmma16(pa, loadB(Vb + ((size_t)(16 + r) << 12) + mb + bo16), o1);
      o2 = wmma16(pa, loadB(Vb + ((size_t)(32 + r) << 12) + mb + bo16), o2);
      o3 = wmma16(pa, loadB(Vb + ((size_t)(48 + r) << 12) + mb + bo16), o3);
    }
  }
  // ---- epilogue: vec[b][o][n] = O[n][o] / l[n] ----
  float inv[8];
#pragma unroll
  for (int j = 0; j < 8; ++j) inv[j] = 1.f / lrow[j];
  {
    float* base = vec + ((size_t)b << 18) + n0 + ho;
    v8f os[4] = {o0, o1, o2, o3};
#pragma unroll
    for (int ot = 0; ot < 4; ++ot) {
      float* p = base + (((size_t)(ot * 16 + r)) << 12);
      float4 lo = make_float4(os[ot][0] * inv[0], os[ot][1] * inv[1],
                              os[ot][2] * inv[2], os[ot][3] * inv[3]);
      float4 hi = make_float4(os[ot][4] * inv[4], os[ot][5] * inv[5],
                              os[ot][6] * inv[6], os[ot][7] * inv[7]);
      *(float4*)p = lo;
      *(float4*)(p + 4) = hi;
    }
  }
}

// ---------------------------------------------------------------------------
// Kernel 3: out = x + Wl @ vec + bl   (1x1 conv + residual)
// ---------------------------------------------------------------------------
__global__ __launch_bounds__(256) void out_kernel(
    const float* __restrict__ x, const float* __restrict__ Wl,
    const float* __restrict__ bl, const float* __restrict__ vec,
    float* __restrict__ out)
{
  __shared__ __align__(16) half_t vT[64 * 72];   // [n][o]
  const int b  = blockIdx.x >> 6;
  const int n0 = (blockIdx.x & 63) << 6;
  const float* vb = vec + ((size_t)b << 18);
  const int t = threadIdx.x;
#pragma unroll
  for (int it = 0; it < 16; ++it) {
    int idx = t + it * 256;
    int c = idx >> 6, n = idx & 63;
    vT[n * 72 + c] = (half_t)vb[((size_t)c << 12) + n0 + n];
  }
  __syncthreads();

  const int wave = t >> 5, lane = t & 31;
  const int r    = lane & 15;
  const int ho   = (lane < 16) ? 0 : 8;
  const int bo16 = (lane < 16) ? 0 : 16;
  const half_t* arow = &vT[(wave * 16 + r) * 72];
  const v16h a0 = loadA(arow, ho);
  const v16h a1 = loadA(arow + 32, ho);
#pragma unroll
  for (int it4 = 0; it4 < 4; ++it4) {
    const int c = it4 * 16 + r;                  // output channel i
    v16h b0 = loadB_f32(Wl + c * 64 + bo16);         // B[k=o][col=i] = Wl[i][o]
    v16h b1 = loadB_f32(Wl + c * 64 + 32 + bo16);
    v8f acc = {};
    acc = wmma16(a0, b0, acc);
    acc = wmma16(a1, b1, acc);
    const float bia = bl[c];
    const size_t off = ((size_t)b << 18) + ((size_t)c << 12) + n0 + wave * 16 + ho;
    const float* xp = x + off;
    float* op = out + off;
    float4 xlo = *(const float4*)xp;
    float4 xhi = *(const float4*)(xp + 4);
    float4 rlo = make_float4(xlo.x + acc[0] + bia, xlo.y + acc[1] + bia,
                             xlo.z + acc[2] + bia, xlo.w + acc[3] + bia);
    float4 rhi = make_float4(xhi.x + acc[4] + bia, xhi.y + acc[5] + bia,
                             xhi.z + acc[6] + bia, xhi.w + acc[7] + bia);
    *(float4*)op = rlo;
    *(float4*)(op + 4) = rhi;
  }
}

// ---------------------------------------------------------------------------
extern "C" void kernel_launch(void* const* d_in, const int* in_sizes, int n_in,
                              void* d_out, int out_size, void* d_ws, size_t ws_size,
                              hipStream_t stream) {
  const float* x  = (const float*)d_in[0];
  const float* Wk = (const float*)d_in[1];
  const float* bk = (const float*)d_in[2];
  const float* Wq = (const float*)d_in[3];
  const float* bq = (const float*)d_in[4];
  const float* Wv = (const float*)d_in[5];
  const float* bv = (const float*)d_in[6];
  const float* Wl = (const float*)d_in[7];
  const float* bl = (const float*)d_in[8];
  float* out = (float*)d_out;

  // Workspace layout: Kt | Qt | V (f16, 4MB each) | vec (f32, 8MB) = 20MB.
  char* ws = (char*)d_ws;
  const size_t halfBytes = (size_t)8 * 4096 * 64 * sizeof(half_t);
  half_t* Kt  = (half_t*)(ws);
  half_t* Qt  = (half_t*)(ws + halfBytes);
  half_t* Vn  = (half_t*)(ws + 2 * halfBytes);
  float*  vec = (float*)(ws + 3 * halfBytes);

  proj_kernel<<<512, 256, 0, stream>>>(x, Wk, bk, Wq, bq, Wv, bv, Kt, Qt, Vn);
  attn_kernel<<<256, 256, 0, stream>>>(Kt, Qt, Vn, vec);
  out_kernel<<<512, 256, 0, stream>>>(x, Wl, bl, vec, out);
}
